// Head_21638045237283
// MI455X (gfx1250) — matrix operand order repaired and checked by reading, not compiled, not run on hardware
//
#include <hip/hip_runtime.h>

#define B_ 8
#define T_ 2048
#define C_ 1024
#define H_ 128

#ifndef __has_builtin
#define __has_builtin(x) 0
#endif
#if __has_builtin(__builtin_amdgcn_global_load_async_to_lds_b128) && \
    __has_builtin(__builtin_amdgcn_s_wait_asynccnt)
#define USE_ASYNC 1
#else
#define USE_ASYNC 0
#endif

typedef __attribute__((ext_vector_type(16))) _Float16 v16h;
typedef __attribute__((ext_vector_type(8)))  _Float16 v8h;
typedef __attribute__((ext_vector_type(4)))  _Float16 v4h;
typedef __attribute__((ext_vector_type(8)))  float    v8f;
typedef __attribute__((ext_vector_type(4)))  float    v4f;
typedef __attribute__((ext_vector_type(4)))  int      v4i;

// Exact pointer types for the async-LDS builtin (param0: v4i addrspace(1)*).
typedef __attribute__((address_space(1))) v4i as1_v4i;
typedef __attribute__((address_space(3))) v4i as3_v4i;

__device__ __forceinline__ v8f wmma16(v16h a, v16h b, v8f c) {
  // D = A(16x32 f16) * B(32x16 f16) + C(16x16 f32)
  return __builtin_amdgcn_wmma_f32_16x16x32_f16(
      false, a, false, b, (short)0, c, false, false);
}

// 16-bit operand loader, contiguous k order used here for A and B: lane%16 = row/col, e -> K = (lane/16)*16 + e
// => 16 contiguous halfs per lane starting at p.
__device__ __forceinline__ v16h ld_b16(const _Float16* p) {
  v8h lo = *(const v8h*)p;
  v8h hi = *(const v8h*)(p + 8);
  return __builtin_shufflevector(lo, hi, 0,1,2,3,4,5,6,7,8,9,10,11,12,13,14,15);
}

// A-operand (16x32, 16-bit): lane%16 = row m; e<8 -> K=(lane/16)*8+e,
// e>=8 -> K=16+(lane/16)*8+(e-8). Caller passes p = row_base + (lane/16)*8.
__device__ __forceinline__ v16h ld_a16(const _Float16* p) {
  v8h lo = *(const v8h*)p;
  v8h hi = *(const v8h*)(p + 16);
  return __builtin_shufflevector(lo, hi, 0,1,2,3,4,5,6,7,8,9,10,11,12,13,14,15);
}

// 16B global -> LDS copy: async DMA (ASYNCcnt) when available, else synchronous.
__device__ __forceinline__ void cp_async16(const _Float16* g, _Float16* l) {
#if USE_ASYNC
  __builtin_amdgcn_global_load_async_to_lds_b128(
      (as1_v4i*)g, (as3_v4i*)l, 0, 0);
#else
  *(v8h*)l = *(const v8h*)g;
#endif
}

__device__ __forceinline__ void async_wait(int prefetch_in_flight) {
#if USE_ASYNC
  if (prefetch_in_flight) __builtin_amdgcn_s_wait_asynccnt(32);
  else                    __builtin_amdgcn_s_wait_asynccnt(0);
#endif
  asm volatile("" ::: "memory");
}

// Stage one 32-key block of K (row-major, contiguous 8KB) and V^T (128 rows x 64B)
// into LDS. 32 x b128 transfers per block.
__device__ __forceinline__ void stage_block(const _Float16* kg, const _Float16* vg,
                                            _Float16* kl, _Float16* vl,
                                            int sbase, int lane) {
  const _Float16* ks = kg + (size_t)sbase * H_;
#pragma unroll
  for (int i = 0; i < 16; ++i) {
    int off = i * 256 + lane * 8;                 // halfs
    cp_async16(ks + off, kl + off);
  }
  const _Float16* vs = vg + sbase;
#pragma unroll
  for (int i = 0; i < 16; ++i) {
    int idx = i * 256 + lane * 8;                 // halfs within 32x128 tile
    int h = idx >> 5, o = idx & 31;               // vbuf[h][key] layout
    cp_async16(vs + (size_t)h * T_ + o, vl + idx);
  }
}

// ---------- Prep: W[3][C][H] f32 -> Wt[3][H][C] f16 (transposed, contiguous cols)
__global__ void __launch_bounds__(256)
prep_w(const float* __restrict__ Wk, const float* __restrict__ Wq,
       const float* __restrict__ Wv, _Float16* __restrict__ wt) {
  const int w = blockIdx.x >> 7;                  // 0=K,1=Q,2=V
  const int n = blockIdx.x & (H_ - 1);
  const float* W = (w == 0) ? Wk : (w == 1) ? Wq : Wv;
  _Float16* dst = wt + ((size_t)w * H_ + n) * C_;
  const int k0 = threadIdx.x * 4;
  v4h t;
#pragma unroll
  for (int i = 0; i < 4; ++i) t[i] = (_Float16)W[(size_t)(k0 + i) * H_ + n];
  *(v4h*)(dst + k0) = t;
}

// ---------- Projection: one wave computes full H (8 N-tiles) for a 16-row M tile
// Software-pipelined (ping-pong over K) so each 8-WMMA chain covers the latency
// of the other half-step's operand loads.
__global__ void __launch_bounds__(32)
proj_qkv(const float* __restrict__ x, const _Float16* __restrict__ wt,
         _Float16* __restrict__ kh,   // [B*T, H]
         _Float16* __restrict__ qh,   // [B*T, H]
         _Float16* __restrict__ vt) { // [B, H, T]
  const int lane = threadIdx.x, ln = lane & 15, hf = lane >> 4;
  const int m0 = blockIdx.x * 16;
  const int w  = blockIdx.y;                      // 0=K,1=Q,2=V
  const _Float16* wm = wt + (size_t)w * H_ * C_ + (size_t)ln * C_;
  const float* xrow = x + (size_t)(m0 + ln) * C_;

  auto load_a = [&](int kk) -> v16h {
    const float* pa = xrow + kk + hf * 8;
    v4f a0 = *(const v4f*)(pa);
    v4f a1 = *(const v4f*)(pa + 4);
    v4f a2 = *(const v4f*)(pa + 16);
    v4f a3 = *(const v4f*)(pa + 20);
    v16h a;
#pragma unroll
    for (int i = 0; i < 4; ++i) {
      a[i]      = (_Float16)a0[i];
      a[4 + i]  = (_Float16)a1[i];
      a[8 + i]  = (_Float16)a2[i];
      a[12 + i] = (_Float16)a3[i];
    }
    return a;
  };
  auto load_b = [&](int kk, v16h* bv) {
#pragma unroll
    for (int nt = 0; nt < 8; ++nt)
      bv[nt] = ld_a16(wm + (size_t)nt * 16 * C_ + kk + hf * 8);
  };

  v8f acc[8];
#pragma unroll
  for (int nt = 0; nt < 8; ++nt) acc[nt] = (v8f){};

  v16h aA, aB, bA[8], bB[8];
  aA = load_a(0);
  load_b(0, bA);

  for (int kk = 0; kk < C_; kk += 64) {
    // issue loads for half-step kk+32 before consuming set A
    aB = load_a(kk + 32);
    load_b(kk + 32, bB);
#pragma unroll
    for (int nt = 0; nt < 8; ++nt) acc[nt] = wmma16(aA, bA[nt], acc[nt]);
    // issue loads for half-step kk+64 before consuming set B
    if (kk + 64 < C_) {
      aA = load_a(kk + 64);
      load_b(kk + 64, bA);
    }
#pragma unroll
    for (int nt = 0; nt < 8; ++nt) acc[nt] = wmma16(aB, bB[nt], acc[nt]);
  }

  // C/D layout: VGPR r -> row m0 + r + 8*hf, col nt*16 + ln
  if (w == 2) {
    const int bb = m0 / T_, tb = m0 % T_;
    _Float16* vb = vt + (size_t)bb * H_ * T_;
#pragma unroll
    for (int nt = 0; nt < 8; ++nt)
#pragma unroll
      for (int r = 0; r < 8; ++r)
        vb[(size_t)(nt * 16 + ln) * T_ + tb + r + 8 * hf] = (_Float16)acc[nt][r];
  } else {
    _Float16* dst = ((w == 0) ? kh : qh) + (size_t)m0 * H_;
#pragma unroll
    for (int nt = 0; nt < 8; ++nt)
#pragma unroll
      for (int r = 0; r < 8; ++r)
        dst[(size_t)(r + 8 * hf) * H_ + nt * 16 + ln] = (_Float16)acc[nt][r];
  }
}

// ---------- Flash attention: 1 wave / 16-query tile, async double-buffered K/V
__global__ void __launch_bounds__(32)
flash_attn(const _Float16* __restrict__ qh, const _Float16* __restrict__ kh,
           const _Float16* __restrict__ vt, float* __restrict__ out) {
  const int lane = threadIdx.x, ln = lane & 15, hf = lane >> 4;
  const int qt = blockIdx.x, b = blockIdx.y;
  const int q0 = qt * 16;

  __shared__ __align__(16) _Float16 kbuf[2][32 * H_];  // [key][h]
  __shared__ __align__(16) _Float16 vbuf[2][H_ * 32];  // [h][key]
  __shared__ __align__(16) _Float16 pl[16 * 32];       // P transpose staging

  const _Float16* kg = kh + (size_t)b * T_ * H_;
  const _Float16* vg = vt + (size_t)b * H_ * T_;

  // Q as resident A-operand: 4 chunks of K=32 over H=128
  const _Float16* qrow = qh + ((size_t)b * T_ + q0 + ln) * H_;
  v16h qa[4];
#pragma unroll
  for (int c = 0; c < 4; ++c) qa[c] = ld_b16(qrow + c * 32 + hf * 16);

  float m_st[8], l_st[8], alpha[8];
  v8f oacc[8];
#pragma unroll
  for (int r = 0; r < 8; ++r) { m_st[r] = -__builtin_inff(); l_st[r] = 0.f; }
#pragma unroll
  for (int c = 0; c < 8; ++c) oacc[c] = (v8f){};

  const float scale = 0.08838834764831845f;  // 1/sqrt(H)
  const int nblocks = (qt >> 1) + 1;         // causal: keys [0, q0+16)

  stage_block(kg, vg, kbuf[0], vbuf[0], 0, lane);

  for (int kb = 0; kb < nblocks; ++kb) {
    const int bi = kb & 1;
    const int sbase = kb * 32;
    const int pref = (kb + 1 < nblocks);
    if (pref) stage_block(kg, vg, kbuf[bi ^ 1], vbuf[bi ^ 1], sbase + 32, lane);
    async_wait(pref);   // current block's 32 async ops retired (in-order counter)

    const _Float16* kl = kbuf[bi];
    const _Float16* vl = vbuf[bi];

    // S = Q @ K^T (two 16-key tiles), operands from LDS
    v8f s0 = (v8f){}, s1 = (v8f){};
#pragma unroll
    for (int c = 0; c < 4; ++c) {
      v16h b0 = ld_b16(kl + (size_t)ln * H_ + c * 32 + hf * 16);
      v16h b1 = ld_b16(kl + (size_t)(16 + ln) * H_ + c * 32 + hf * 16);
      s0 = wmma16(qa[c], b0, s0);
      s1 = wmma16(qa[c], b1, s1);
    }
#pragma unroll
    for (int r = 0; r < 8; ++r) { s0[r] *= scale; s1[r] *= scale; }

    if (kb == nblocks - 1) { // only the diagonal block needs masking
#pragma unroll
      for (int r = 0; r < 8; ++r) {
        int qg = q0 + r + 8 * hf;
        if (sbase + ln      > qg) s0[r] = -__builtin_inff();
        if (sbase + 16 + ln > qg) s1[r] = -__builtin_inff();
      }
    }

    // Online softmax; row m = r + 8*hf lives across lanes [hf*16, hf*16+15]
#pragma unroll
    for (int r = 0; r < 8; ++r) {
      float v = fmaxf(s0[r], s1[r]);
      v = fmaxf(v, __shfl_xor(v, 1));
      v = fmaxf(v, __shfl_xor(v, 2));
      v = fmaxf(v, __shfl_xor(v, 4));
      v = fmaxf(v, __shfl_xor(v, 8));
      float mnew = fmaxf(m_st[r], v);
      alpha[r] = __expf(m_st[r] - mnew);
      m_st[r] = mnew;
      float p0 = __expf(s0[r] - mnew);
      float p1 = __expf(s1[r] - mnew);
      s0[r] = p0; s1[r] = p1;
      float rs = p0 + p1;
      rs += __shfl_xor(rs, 1);
      rs += __shfl_xor(rs, 2);
      rs += __shfl_xor(rs, 4);
      rs += __shfl_xor(rs, 8);
      l_st[r] = l_st[r] * alpha[r] + rs;
    }

#pragma unroll
    for (int c = 0; c < 8; ++c)
#pragma unroll
      for (int r = 0; r < 8; ++r) oacc[c][r] *= alpha[r];

    // P: C-layout -> A-layout via LDS (per-wave DS ops are in-order)
    __syncthreads();
#pragma unroll
    for (int r = 0; r < 8; ++r) {
      int m = r + 8 * hf;
      pl[m * 32 + ln]      = (_Float16)s0[r];
      pl[m * 32 + 16 + ln] = (_Float16)s1[r];
    }
    __syncthreads();
    v16h pa = ld_b16(pl + ln * 32 + hf * 16);

    // O += P(16x32) @ V(32x128), V from LDS [h][key] => contiguous keys
#pragma unroll
    for (int c = 0; c < 8; ++c) {
      v16h vtile = ld_b16(vl + (size_t)(c * 16 + ln) * 32 + hf * 16);
      oacc[c] = wmma16(pa, vtile, oacc[c]);
    }
  }

  // Epilogue: out = O / l
#pragma unroll
  for (int c = 0; c < 8; ++c)
#pragma unroll
    for (int r = 0; r < 8; ++r) {
      int m = q0 + r + 8 * hf;
      out[((size_t)b * T_ + m) * H_ + c * 16 + ln] = oacc[c][r] / l_st[r];
    }
}

extern "C" void kernel_launch(void* const* d_in, const int* in_sizes, int n_in,
                              void* d_out, int out_size, void* d_ws, size_t ws_size,
                              hipStream_t stream) {
  (void)in_sizes; (void)n_in; (void)out_size; (void)ws_size;
  const float* x  = (const float*)d_in[0];
  const float* Wk = (const float*)d_in[1];
  const float* Wq = (const float*)d_in[2];
  const float* Wv = (const float*)d_in[3];

  const size_t elems = (size_t)B_ * T_ * H_;     // 2M halfs per tensor
  _Float16* kh = (_Float16*)d_ws;
  _Float16* qh = kh + elems;
  _Float16* vt = qh + elems;
  _Float16* wt = vt + elems;                     // 3*H*C halfs = 0.75 MB
  float* out = (float*)d_out;

  prep_w<<<3 * H_, 256, 0, stream>>>(Wk, Wq, Wv, wt);

  dim3 g1((B_ * T_) / 16, 3);                    // 1024 M-tiles x 3 matrices
  proj_qkv<<<g1, 32, 0, stream>>>(x, wt, kh, qh, vt);

  dim3 g2(T_ / 16, B_);                          // 128 query tiles x 8 batches
  flash_attn<<<g2, 32, 0, stream>>>(qh, kh, vt, out);
}
